// Attention_37263136260865
// MI455X (gfx1250) — matrix operand order repaired and checked by reading, not compiled, not run on hardware
//
#include <hip/hip_runtime.h>

typedef __attribute__((ext_vector_type(2)))  float    v2f;
typedef __attribute__((ext_vector_type(8)))  float    v8f;
typedef __attribute__((ext_vector_type(16))) _Float16 v16h;

union V16 {
    v16h  v;
    _Float16 h[16];
    unsigned int u[8];
    uint4 q[2];
};

#define CH  64
#define B_  2
#define N_  32768
#define M_  4096
#define KB_ (M_ / 32)   // 128 key-blocks of 32

// ---------------------------------------------------------------------------
// Kernel 1: channel projections. theta -> Qh[B,N,8] (f16, query-major),
// phi -> Pf[B,N,8] (f32), g -> Gf[B,N,32] (f32). Coalesced on n.
// ---------------------------------------------------------------------------
__global__ __launch_bounds__(256) void proj_kernel(
    const float* __restrict__ x,  const float* __restrict__ wt,
    const float* __restrict__ wp, const float* __restrict__ wg,
    _Float16* __restrict__ Qh, float* __restrict__ Pf, float* __restrict__ Gf)
{
    int idx = blockIdx.x * 256 + threadIdx.x;   // b*N + n
    int b = idx >> 15;
    int n = idx & (N_ - 1);
    const float* xp = x + (size_t)b * CH * N_ + n;

    float at[8] = {}, ap[8] = {}, ag[32] = {};
    #pragma unroll 4
    for (int c = 0; c < CH; ++c) {
        float xv = xp[(size_t)c * N_];
        #pragma unroll
        for (int o = 0; o < 8; ++o)  at[o] = fmaf(wt[o * CH + c], xv, at[o]);
        #pragma unroll
        for (int o = 0; o < 8; ++o)  ap[o] = fmaf(wp[o * CH + c], xv, ap[o]);
        #pragma unroll
        for (int o = 0; o < 32; ++o) ag[o] = fmaf(wg[o * CH + c], xv, ag[o]);
    }
    _Float16* q = Qh + (size_t)idx * 8;
    float*    p = Pf + (size_t)idx * 8;
    float*    g = Gf + (size_t)idx * 32;
    #pragma unroll
    for (int o = 0; o < 8; ++o)  q[o] = (_Float16)at[o];
    #pragma unroll
    for (int o = 0; o < 8; ++o)  p[o] = ap[o];
    #pragma unroll
    for (int o = 0; o < 32; ++o) g[o] = ag[o];
}

// ---------------------------------------------------------------------------
// Kernel 2: 2x2x2 max-pool.
//   phi_full -> Kh[B,M,8] (f16 rows, 16B aligned for b128 tile loads)
//   g_full   -> Vpack: per (b, key-block, ch-half) the exact 32x16 f16
//               B-tile register image: dword[lane*8+j] = pack(V[2j+16*hi][c],
//               V[2j+1+16*hi][c]) with lane = hi*16 + (c&15).
// ---------------------------------------------------------------------------
__global__ __launch_bounds__(256) void pool_kernel(
    const float* __restrict__ Pf, const float* __restrict__ Gf,
    _Float16* __restrict__ Kh, unsigned int* __restrict__ Vpack)
{
    int t = blockIdx.x * 256 + threadIdx.x;
    const int phiTot = B_ * M_ * 8;
    if (t < phiTot) {
        int c = t & 7; int bm = t >> 3;
        int b = bm >> 12; int m = bm & (M_ - 1);
        int wx = m & 15, hy = (m >> 4) & 15, dz = m >> 8;
        float mx = -__builtin_inff();
        #pragma unroll
        for (int od = 0; od < 2; ++od)
        #pragma unroll
        for (int oh = 0; oh < 2; ++oh)
        #pragma unroll
        for (int ow = 0; ow < 2; ++ow) {
            int n = (2 * dz + od) * 1024 + (2 * hy + oh) * 32 + (2 * wx + ow);
            mx = fmaxf(mx, Pf[((size_t)b * N_ + n) * 8 + c]);
        }
        Kh[((size_t)b * M_ + m) * 8 + c] = (_Float16)mx;
    } else {
        t -= phiTot;
        int c = t & 31; int bm = t >> 5;
        int b = bm >> 12; int m = bm & (M_ - 1);
        int wx = m & 15, hy = (m >> 4) & 15, dz = m >> 8;
        float mx = -__builtin_inff();
        #pragma unroll
        for (int od = 0; od < 2; ++od)
        #pragma unroll
        for (int oh = 0; oh < 2; ++oh)
        #pragma unroll
        for (int ow = 0; ow < 2; ++ow) {
            int n = (2 * dz + od) * 1024 + (2 * hy + oh) * 32 + (2 * wx + ow);
            mx = fmaxf(mx, Gf[((size_t)b * N_ + n) * 32 + c]);
        }
        // scatter into B-tile image
        int kb = m >> 5, kk = m & 31;
        int hi = kk >> 4, j = (kk & 15) >> 1, hp = kk & 1;
        int h = c >> 4, l16 = c & 15;
        int lane = hi * 16 + l16;
        size_t dw = (((size_t)b * KB_ + kb) * 2 + h) * 256 + lane * 8 + j;
        ((_Float16*)Vpack)[dw * 2 + hp] = (_Float16)mx;
    }
}

// ---------------------------------------------------------------------------
// Kernel 3: streaming attention, f16 WMMA 16x16x32, f32 accumulate.
// 8 waves/block; each wave owns 16 queries, streams M keys in 32-key blocks.
// Fixed per-row exp offset = rowmax of first 16-key tile (computed once).
// Row sums accumulated with an all-ones WMMA; P bounced via f16 LDS tile so
// A-tile dwords are single aligned ds_load_b32 of pre-packed f16 pairs.
// Output written as f16 [pos][32] for the WMMA epilogue kernel.
// ---------------------------------------------------------------------------
__global__ __launch_bounds__(256) void flash_kernel(
    const _Float16* __restrict__ Qh, const _Float16* __restrict__ Kh,
    const unsigned int* __restrict__ Vpack, _Float16* __restrict__ Obh)
{
    __shared__ _Float16 sP[8 * 16 * 32];   // 16 rows x 32 keys of P per wave

    const int lane = threadIdx.x & 31;
    const int wave = threadIdx.x >> 5;
    const int half = lane >> 4;          // C-layout: rows r (0) / r+8 (1)
    const int l16  = lane & 15;
    const int bpb  = N_ / 128;
    const int b     = blockIdx.x / bpb;
    const int qbase = (blockIdx.x % bpb) * 128 + wave * 16;

    const uint4 z4 = {0u, 0u, 0u, 0u};

    // Q tile, A layout: lanes 0-15 carry dims 0-7 of row l16; rest zero-pad.
    V16 qa; qa.q[0] = z4; qa.q[1] = z4;
    if (half == 0)
        qa.q[0] = *(const uint4*)(Qh + ((size_t)b * N_ + qbase + l16) * 8);

    V16 onesB;
    #pragma unroll
    for (int i = 0; i < 16; ++i) onesB.h[i] = (_Float16)1.0f;

    const _Float16* Kb = Kh + (size_t)b * M_ * 8;
    const unsigned int* Vb = Vpack + (size_t)b * KB_ * 2 * 256;

    // ---- one-time per-row exp offset from the first 16-key tile ----
    float mrow[8];
    {
        V16 kt; kt.q[0] = z4; kt.q[1] = z4;
        if (half == 0)
            kt.q[0] = *(const uint4*)(Kb + (size_t)l16 * 8);
        v8f s = {};
        s = __builtin_amdgcn_wmma_f32_16x16x32_f16(false, qa.v, false, kt.v,
                                                   (short)0, s, false, false);
        #pragma unroll
        for (int r = 0; r < 8; ++r) {
            float mx = s[r];
            mx = fmaxf(mx, __shfl_xor(mx, 1, 32));
            mx = fmaxf(mx, __shfl_xor(mx, 2, 32));
            mx = fmaxf(mx, __shfl_xor(mx, 4, 32));
            mx = fmaxf(mx, __shfl_xor(mx, 8, 32));
            mrow[r] = mx;
        }
    }

    v8f o0 = {}, o1 = {}, lacc = {};
    _Float16* myP = sP + wave * 512;

    for (int kb = 0; kb < KB_; ++kb) {
        const int k0 = kb * 32;
        // Two K B-tiles (16 keys each): lanes 0-15 carry dims 0-7 of key l16.
        V16 kb0, kb1;
        kb0.q[0] = z4; kb0.q[1] = z4; kb1.q[0] = z4; kb1.q[1] = z4;
        if (half == 0) {
            kb0.q[0] = *(const uint4*)(Kb + (size_t)(k0 + l16) * 8);
            kb1.q[0] = *(const uint4*)(Kb + (size_t)(k0 + 16 + l16) * 8);
        }
        if (kb + 1 < KB_) {
            __builtin_prefetch(Kb + (size_t)(k0 + 32 + l16) * 8, 0, 1);
            __builtin_prefetch(Vb + ((size_t)(kb + 1) * 2) * 256 + lane * 8, 0, 1);
        }

        v8f s0 = {}, s1 = {};
        s0 = __builtin_amdgcn_wmma_f32_16x16x32_f16(false, qa.v, false, kb0.v,
                                                    (short)0, s0, false, false);
        s1 = __builtin_amdgcn_wmma_f32_16x16x32_f16(false, qa.v, false, kb1.v,
                                                    (short)0, s1, false, false);

        // P = exp(S - mrow), converted to f16 and stored to the LDS tile.
        #pragma unroll
        for (int r = 0; r < 8; ++r) {
            float p0 = __expf(s0[r] - mrow[r]);
            float p1 = __expf(s1[r] - mrow[r]);
            myP[(r + 8 * half) * 32 + l16]      = (_Float16)p0;
            myP[(r + 8 * half) * 32 + 16 + l16] = (_Float16)p1;
        }
        asm volatile("s_wait_dscnt 0" ::: "memory");

        // P tile in A layout: dword j -> keys 16*half+{2j,2j+1};
        // each one aligned b32 LDS load.
        const unsigned int* prow = (const unsigned int*)(myP + l16 * 32);
        V16 pa;
        #pragma unroll
        for (int j = 0; j < 8; ++j) pa.u[j]     = prow[8 * half + j];

        // V B-tiles: prepacked register images, two b128 loads each.
        const unsigned int* t0 = Vb + ((size_t)kb * 2 + 0) * 256 + lane * 8;
        const unsigned int* t1 = Vb + ((size_t)kb * 2 + 1) * 256 + lane * 8;
        V16 vb0, vb1;
        vb0.q[0] = *(const uint4*)(t0);
        vb0.q[1] = *(const uint4*)(t0 + 4);
        vb1.q[0] = *(const uint4*)(t1);
        vb1.q[1] = *(const uint4*)(t1 + 4);

        o0   = __builtin_amdgcn_wmma_f32_16x16x32_f16(false, pa.v, false, vb0.v,
                                                      (short)0, o0,   false, false);
        o1   = __builtin_amdgcn_wmma_f32_16x16x32_f16(false, pa.v, false, vb1.v,
                                                      (short)0, o1,   false, false);
        lacc = __builtin_amdgcn_wmma_f32_16x16x32_f16(false, pa.v, false, onesB.v,
                                                      (short)0, lacc, false, false);
    }

    // Normalize and store as f16 [pos][32]; exp offset cancels in o/l.
    #pragma unroll
    for (int r = 0; r < 8; ++r) {
        float inv = 1.0f / lacc[r];
        size_t row = (size_t)b * N_ + qbase + r + 8 * half;
        _Float16* orow = Obh + row * 32;
        orow[l16]      = (_Float16)(o0[r] * inv);
        orow[16 + l16] = (_Float16)(o1[r] * inv);
    }
}

// ---------------------------------------------------------------------------
// Kernel 4: out = gamma * (W_o [64x32] . o) + x, via WMMA.
// Each wave holds 4 persistent W_o A-tiles (16 out-ch x 32 value-ch) and
// processes 16-position tiles: B-tile = 8 aligned b32 loads of f16 pairs.
// ---------------------------------------------------------------------------
#define OT_TILES (B_ * N_ / 16)   // 4096 position tiles
__global__ __launch_bounds__(256) void out_kernel(
    const float* __restrict__ x, const float* __restrict__ wo,
    const float* __restrict__ gamma, const _Float16* __restrict__ Obh,
    float* __restrict__ out)
{
    const int lane = threadIdx.x & 31;
    const int wave = threadIdx.x >> 5;
    const int hi   = lane >> 4;
    const int l16  = lane & 15;
    const int waveId = blockIdx.x * 8 + wave;       // 1024 waves
    const int TPW = OT_TILES / 1024;                // 4 tiles per wave

    // Build 4 A-tiles of W_o: rows t*16+l16, K = 32 value channels.
    V16 a[4];
    #pragma unroll
    for (int t = 0; t < 4; ++t) {
        const float* wr = wo + (size_t)(t * 16 + l16) * 32;
        #pragma unroll
        for (int j = 0; j < 8; ++j) {
            int kb = 16 * hi + 2 * j;
            a[t].h[2 * j]     = (_Float16)wr[kb];
            a[t].h[2 * j + 1] = (_Float16)wr[kb + 1];
        }
    }
    float gm = gamma[0];

    for (int it = 0; it < TPW; ++it) {
        int tile = waveId * TPW + it;
        int n0 = tile * 16;
        int b  = n0 >> 15;
        int nn = (n0 & (N_ - 1)) + l16;
        int pos = n0 + l16;

        // B-tile (32 value-ch x 16 positions): dword j = ch pair {2j+16*hi}.
        const unsigned int* orow = (const unsigned int*)(Obh + (size_t)pos * 32);
        V16 bv;
        #pragma unroll
        for (int j = 0; j < 8; ++j) bv.u[j] = orow[j + 8 * hi];

        const float* xb = x   + (size_t)b * CH * N_;
        float*       yb = out + (size_t)b * CH * N_;
        #pragma unroll
        for (int t = 0; t < 4; ++t) {
            v8f z = {};
            v8f d = __builtin_amdgcn_wmma_f32_16x16x32_f16(false, a[t].v, false, bv.v,
                                                           (short)0, z, false, false);
            #pragma unroll
            for (int r = 0; r < 8; ++r) {
                int c = t * 16 + r + 8 * hi;
                yb[(size_t)c * N_ + nn] = fmaf(gm, d[r], xb[(size_t)c * N_ + nn]);
            }
        }
    }
}

// ---------------------------------------------------------------------------
extern "C" void kernel_launch(void* const* d_in, const int* in_sizes, int n_in,
                              void* d_out, int out_size, void* d_ws, size_t ws_size,
                              hipStream_t stream)
{
    const float* x     = (const float*)d_in[0];
    const float* wt    = (const float*)d_in[1];
    const float* wp    = (const float*)d_in[2];
    const float* wg    = (const float*)d_in[3];
    const float* wo    = (const float*)d_in[4];
    const float* gamma = (const float*)d_in[5];
    float* out = (float*)d_out;

    float* Pf = (float*)d_ws;                               // B*N*8   f32
    float* Gf = Pf + (size_t)B_ * N_ * 8;                   // B*N*32  f32
    _Float16* Qh  = (_Float16*)(Gf + (size_t)B_ * N_ * 32); // B*N*8   f16
    _Float16* Kh  = Qh + (size_t)B_ * N_ * 8;               // B*M*8   f16
    _Float16* Obh = Kh + (size_t)B_ * M_ * 8;               // B*N*32  f16
    unsigned int* Vpack = (unsigned int*)(Obh + (size_t)B_ * N_ * 32); // B*KB*2*256

    proj_kernel <<<B_ * N_ / 256,      256, 0, stream>>>(x, wt, wp, wg, Qh, Pf, Gf);
    pool_kernel <<<B_ * M_ * 40 / 256, 256, 0, stream>>>(Pf, Gf, Kh, Vpack);
    flash_kernel<<<B_ * N_ / 128,      256, 0, stream>>>(Qh, Kh, Vpack, Obh);
    out_kernel  <<<OT_TILES / (8 * (OT_TILES / 1024)), 256, 0, stream>>>(x, wo, gamma, Obh, out);
}
